// AdaptiveConv3_31799937860209
// MI455X (gfx1250) — matrix operand order repaired and checked by reading, not compiled, not run on hardware
//
#include <hip/hip_runtime.h>

// ---------------------------------------------------------------------------
// Problem constants
// ---------------------------------------------------------------------------
#define NB    8      // batch
#define CIN   64     // in channels (== INTER for all conv layers)
#define HW    128    // H == W
#define GOUT  36     // generator out channels
#define NLAYER 7     // conv0 + 5 mid + final
#define KSTEPS 18    // K = 9 taps * 64 ic -> 18 chunks of 32 (bf16 WMMA K)
#define APACK_PER_LAYER (KSTEPS * 4 * 32 * 8)   // dwords: 18432

// ---------------------------------------------------------------------------
// CDNA5 WMMA types
// ---------------------------------------------------------------------------
typedef __bf16 bf16_t;
typedef bf16_t v16bf __attribute__((ext_vector_type(16)));
typedef float  v8f   __attribute__((ext_vector_type(8)));

union Frag { unsigned int u[8]; v16bf v; };

__device__ __forceinline__ unsigned int pack_bf2(float lo, float hi) {
    unsigned int a = __float_as_uint(lo);
    unsigned int b = __float_as_uint(hi);
    unsigned int ra = (a + 0x7fffu + ((a >> 16) & 1u)) >> 16;
    unsigned int rb = (b + 0x7fffu + ((b >> 16) & 1u)) & 0xffff0000u;
    return ra | rb;
}

// ---------------------------------------------------------------------------
// prep_params: fold conv-bias + BN (eval) into per-channel alpha/beta
//   layer 0   : alpha = g0/sqrt(v0+eps), beta = (b0-m0)*alpha + be0, tanh
//   layer 1..5: alpha = 1,               beta = bmid[i-1]
//   layer 6   : alpha = gf/sqrt(vf+eps), beta = (bf-mf)*alpha + bef, tanh
// alpha/beta padded to 64 channels per layer.
// ---------------------------------------------------------------------------
__global__ void prep_params(const float* __restrict__ b0, const float* __restrict__ g0,
                            const float* __restrict__ be0, const float* __restrict__ m0,
                            const float* __restrict__ v0, const float* __restrict__ bmid,
                            const float* __restrict__ bf_, const float* __restrict__ gf,
                            const float* __restrict__ bef, const float* __restrict__ mf,
                            const float* __restrict__ vf,
                            float* __restrict__ alpha, float* __restrict__ beta)
{
    int tid = blockIdx.x * blockDim.x + threadIdx.x;
    if (tid >= NLAYER * 64) return;
    int layer = tid >> 6, oc = tid & 63;
    float a = 0.f, b = 0.f;
    if (layer == 0) {
        float s = g0[oc] * rsqrtf(v0[oc] + 1e-5f);
        a = s; b = (b0[oc] - m0[oc]) * s + be0[oc];
    } else if (layer <= 5) {
        a = 1.f; b = bmid[(layer - 1) * 64 + oc];
    } else if (oc < GOUT) {
        float s = gf[oc] * rsqrtf(vf[oc] + 1e-5f);
        a = s; b = (bf_[oc] - mf[oc]) * s + bef[oc];
    }
    alpha[tid] = a; beta[tid] = b;
}

// ---------------------------------------------------------------------------
// prep_weights: pack fp32 OIHW weights -> bf16 A-fragments in a per-lane
// WMMA layout (16-bit A 16x32, contiguous k = 16h + s, same as B).  Index:
//   Apack[(((layer*18 + s)*4 + mblock)*32 + lane)*8 + v]
// K-step s: tap t = s>>1 (ky*3+kx), ic chunk = (s&1)*32.
// Lane layout: m = lane&15; kbase = lane<16 ? 0 : 16;
//   K = kbase + 2v; pair (K,K+1)
// ---------------------------------------------------------------------------
__global__ void prep_weights(const float* __restrict__ W0,
                             const float* __restrict__ Wmid,
                             const float* __restrict__ Wf,
                             unsigned int* __restrict__ Apack)
{
    int tid = blockIdx.x * blockDim.x + threadIdx.x;
    if (tid >= NLAYER * APACK_PER_LAYER) return;
    int v      =  tid        & 7;
    int lane   = (tid >> 3)  & 31;
    int mblock = (tid >> 8)  & 3;
    int rest   =  tid >> 10;            // layer*18 + s
    int s      = rest % KSTEPS;
    int layer  = rest / KSTEPS;

    int oc    = mblock * 16 + (lane & 15);
    int kbase = (lane < 16) ? 0 : 16;
    int kk    = kbase + 2 * v;
    int icB   = (s & 1) * 32;
    int t     = s >> 1;                 // tap = ky*3+kx
    int ic0   = icB + kk;

    const float* Wsrc; int OC;
    if (layer == 0)      { Wsrc = W0;                          OC = 64;   }
    else if (layer <= 5) { Wsrc = Wmid + (layer - 1) * 64 * 64 * 9; OC = 64; }
    else                 { Wsrc = Wf;                          OC = GOUT; }

    float w0 = 0.f, w1 = 0.f;
    if (oc < OC) {
        w0 = Wsrc[(oc * CIN + ic0)     * 9 + t];
        w1 = Wsrc[(oc * CIN + ic0 + 1) * 9 + t];
    }
    Apack[tid] = pack_bf2(w0, w1);
}

// ---------------------------------------------------------------------------
// nchw_to_nhwc: input fp32 NCHW -> bf16 NHWC, LDS-staged transpose.
// One block per (n,y) row.
// ---------------------------------------------------------------------------
__global__ __launch_bounds__(256)
void nchw_to_nhwc(const float* __restrict__ in, unsigned short* __restrict__ out)
{
    __shared__ float st[HW][65];        // [x][c], pad 65 to kill bank conflicts
    int y = blockIdx.x, n = blockIdx.y;
    int tid = threadIdx.x;
    int x = tid & 127;
    for (int c = (tid >> 7); c < CIN; c += 2)
        st[x][c] = in[((size_t)(n * CIN + c) * HW + y) * HW + x];
    __syncthreads();
    unsigned int* o32 = (unsigned int*)out;
    for (int idx = tid; idx < HW * 32; idx += 256) {
        int xx = idx >> 5, cp = idx & 31;
        o32[((size_t)(n * HW + y) * HW + xx) * 32 + cp] =
            pack_bf2(st[xx][2 * cp], st[xx][2 * cp + 1]);
    }
}

// ---------------------------------------------------------------------------
// conv_wmma: one 3x3 conv layer as implicit GEMM, bf16 WMMA, f32 accumulate.
//   grid = (HW rows, NB), block = 256 (8 waves).
//   WG computes 64 oc x 128 px; wave w: mblock = w&3 (16 oc), nhalf = w>>2
//   (64 px = 4 N-tiles of 16).  K loop = 18 steps of 32 (tap-major).
//   LDS tile: 3 rows x 130 cols x 64 ic bf16 ([row][col][ic]) so each B
//   fragment is 2 contiguous ds_load_b128 per lane.
// ---------------------------------------------------------------------------
__global__ __launch_bounds__(256)
void conv_wmma(const unsigned short* __restrict__ in,   // NHWC bf16
               void* __restrict__ outp,                 // bf16 NHWC or f32 NHWC(36)
               const unsigned int* __restrict__ apackL, // this layer's Apack
               const float* __restrict__ alpha,
               const float* __restrict__ beta,
               int doTanh, int outF32)
{
    __shared__ unsigned short tile[3 * 130 * CIN];      // 49,920 B
    int y = blockIdx.x, n = blockIdx.y;
    int tid = threadIdx.x;

    // ---- stage input rows y-1..y+1 (halo cols -1..128), zero padded ----
    for (int idx = tid; idx < 390 * 8; idx += 256) {    // 390 pixels * 8 uint4
        int pix = idx >> 3, chunk = idx & 7;
        int r = pix / 130, col = pix - r * 130;
        int yy = y + r - 1, xx = col - 1;
        uint4 val = make_uint4(0u, 0u, 0u, 0u);
        if ((unsigned)yy < (unsigned)HW && (unsigned)xx < (unsigned)HW)
            val = reinterpret_cast<const uint4*>(
                      in + ((size_t)((n * HW + yy) * HW + xx) * CIN))[chunk];
        reinterpret_cast<uint4*>(&tile[(r * 130 + col) * CIN])[chunk] = val;
    }
    __syncthreads();

    int wv = tid >> 5, lane = tid & 31;
    int mblock = wv & 3, nhalf = wv >> 2;
    int l15 = lane & 15;
    int kbB = (lane < 16) ? 0 : 16;      // B-fragment K base for this lane

    v8f zero = {0.f, 0.f, 0.f, 0.f, 0.f, 0.f, 0.f, 0.f};
    v8f acc[4];
#pragma unroll
    for (int i = 0; i < 4; ++i) acc[i] = zero;

    for (int s = 0; s < KSTEPS; ++s) {
        int t = s >> 1, icB = (s & 1) * 32;
        int ky = t / 3, kx = t - ky * 3;

        Frag a;
        const uint4* ap = reinterpret_cast<const uint4*>(
            apackL + (size_t)(((s * 4 + mblock) * 32 + lane) * 8));
        *(uint4*)(&a.u[0]) = ap[0];
        *(uint4*)(&a.u[4]) = ap[1];

        int icOff = icB + kbB;
#pragma unroll
        for (int i = 0; i < 4; ++i) {
            int col = nhalf * 64 + i * 16 + l15 + kx;    // tile col (x-1 at col 0)
            const uint4* bp = reinterpret_cast<const uint4*>(
                &tile[(ky * 130 + col) * CIN + icOff]);
            Frag b;
            *(uint4*)(&b.u[0]) = bp[0];
            *(uint4*)(&b.u[4]) = bp[1];
            acc[i] = __builtin_amdgcn_wmma_f32_16x16x32_bf16(
                         false, a.v, false, b.v, (short)0, acc[i], false, false);
        }
    }

    // ---- epilogue: per-channel alpha/beta (+tanh), store NHWC ----
    int ocL = mblock * 16 + ((lane < 16) ? 0 : 8);
    float al[8], bt[8];
#pragma unroll
    for (int r = 0; r < 8; ++r) { al[r] = alpha[ocL + r]; bt[r] = beta[ocL + r]; }

    size_t rowBase = (size_t)(n * HW + y) * HW;
#pragma unroll
    for (int i = 0; i < 4; ++i) {
        int x = nhalf * 64 + i * 16 + l15;
        float v[8];
#pragma unroll
        for (int r = 0; r < 8; ++r) {
            float f = acc[i][r] * al[r] + bt[r];
            v[r] = doTanh ? tanhf(f) : f;
        }
        if (!outF32) {
            unsigned short* o = (unsigned short*)outp;
            uint4 pk = make_uint4(pack_bf2(v[0], v[1]), pack_bf2(v[2], v[3]),
                                  pack_bf2(v[4], v[5]), pack_bf2(v[6], v[7]));
            *reinterpret_cast<uint4*>(o + ((rowBase + x) * CIN + ocL)) = pk;
        } else {
            float* o = (float*)outp;                    // gen: NHWC, 36 channels
            if (ocL < GOUT) {
                size_t base = (rowBase + x) * GOUT + ocL;
                *reinterpret_cast<float4*>(o + base) =
                    make_float4(v[0], v[1], v[2], v[3]);
                if (ocL + 7 < GOUT)
                    *reinterpret_cast<float4*>(o + base + 4) =
                        make_float4(v[4], v[5], v[6], v[7]);
            }
        }
    }
}

// ---------------------------------------------------------------------------
// fused_einsum: kern = gen x bases (per pixel), then per-pixel 3x3 adaptive
// conv against all 64 input channels.  16x16 pixel tile per block; input
// channel tiles (18x18 halo) staged through LDS.  Memory bound on the
// 201 MB output write.
// ---------------------------------------------------------------------------
__global__ __launch_bounds__(256)
void fused_einsum(const float* __restrict__ input,   // fp32 NCHW
                  const float* __restrict__ gen,     // fp32 NHWC [.,.,.,36]
                  const float* __restrict__ bases,   // [6][9]
                  float* __restrict__ out)           // fp32 [N,384,H,W]
{
    __shared__ float sb[54];
    __shared__ float stile[18 * 18];
    int tid = threadIdx.x;
    int lx = tid & 15, ly = tid >> 4;
    int x = blockIdx.x * 16 + lx;
    int y = blockIdx.y * 16 + ly;
    int n = blockIdx.z;

    if (tid < 54) sb[tid] = bases[tid];

    float g[36];
    const float4* gp = reinterpret_cast<const float4*>(
        gen + ((size_t)(n * HW + y) * HW + x) * GOUT);
#pragma unroll
    for (int i = 0; i < 9; ++i) ((float4*)g)[i] = gp[i];
    __syncthreads();

    float kern[54];
#pragma unroll
    for (int f = 0; f < 6; ++f)
#pragma unroll
        for (int d = 0; d < 9; ++d) {
            float s = 0.f;
#pragma unroll
            for (int l = 0; l < 6; ++l) s += g[f * 6 + l] * sb[l * 9 + d];
            kern[f * 9 + d] = s;
        }

    for (int c = 0; c < CIN; ++c) {
        __syncthreads();
        const float* ip = input + (size_t)(n * CIN + c) * HW * HW;
        for (int idx = tid; idx < 324; idx += 256) {
            int r = idx / 18, cc = idx - r * 18;
            int yy = blockIdx.y * 16 + r - 1;
            int xx = blockIdx.x * 16 + cc - 1;
            stile[idx] = ((unsigned)yy < (unsigned)HW && (unsigned)xx < (unsigned)HW)
                             ? ip[yy * HW + xx] : 0.f;
        }
        __syncthreads();
        float nb[9];
#pragma unroll
        for (int dy = 0; dy < 3; ++dy)
#pragma unroll
            for (int dx = 0; dx < 3; ++dx)
                nb[dy * 3 + dx] = stile[(ly + dy) * 18 + (lx + dx)];
#pragma unroll
        for (int m = 0; m < 6; ++m) {
            float s = 0.f;
#pragma unroll
            for (int d = 0; d < 9; ++d) s += kern[m * 9 + d] * nb[d];
            out[(((size_t)n * 384 + (c * 6 + m)) * HW + y) * HW + x] = s;
        }
    }
}

// ---------------------------------------------------------------------------
// kernel_launch
// ---------------------------------------------------------------------------
extern "C" void kernel_launch(void* const* d_in, const int* in_sizes, int n_in,
                              void* d_out, int out_size, void* d_ws, size_t ws_size,
                              hipStream_t stream)
{
    (void)in_sizes; (void)n_in; (void)out_size; (void)ws_size;
    const float* input = (const float*)d_in[0];
    const float* W0    = (const float*)d_in[1];
    const float* b0    = (const float*)d_in[2];
    const float* g0    = (const float*)d_in[3];
    const float* be0   = (const float*)d_in[4];
    const float* m0    = (const float*)d_in[5];
    const float* v0    = (const float*)d_in[6];
    const float* Wmid  = (const float*)d_in[7];
    const float* bmid  = (const float*)d_in[8];
    const float* Wf    = (const float*)d_in[9];
    const float* bf_   = (const float*)d_in[10];
    const float* gf    = (const float*)d_in[11];
    const float* bef   = (const float*)d_in[12];
    const float* mf    = (const float*)d_in[13];
    const float* vf    = (const float*)d_in[14];
    const float* bases = (const float*)d_in[15];

    char* ws = (char*)d_ws;
    const size_t APACK_BYTES = (size_t)NLAYER * APACK_PER_LAYER * 4;   // 516,096
    unsigned int*   Apack = (unsigned int*)ws;
    float*          alpha = (float*)(ws + APACK_BYTES);                // 7*64 f32
    float*          beta  = (float*)(ws + APACK_BYTES + 1792);
    const size_t ACT0 = APACK_BYTES + 3584;                            // 519,680
    const size_t ACT_BYTES = (size_t)NB * HW * HW * CIN * 2;           // 16.78 MB
    unsigned short* actA  = (unsigned short*)(ws + ACT0);
    unsigned short* actB  = (unsigned short*)(ws + ACT0 + ACT_BYTES);
    float*          gen   = (float*)(ws + ACT0 + 2 * ACT_BYTES);      // 18.87 MB

    prep_params<<<2, 256, 0, stream>>>(b0, g0, be0, m0, v0, bmid,
                                       bf_, gf, bef, mf, vf, alpha, beta);
    prep_weights<<<(NLAYER * APACK_PER_LAYER + 255) / 256, 256, 0, stream>>>(
        W0, Wmid, Wf, Apack);
    nchw_to_nhwc<<<dim3(HW, NB), 256, 0, stream>>>(input, actA);

    dim3 cg(HW, NB);
    conv_wmma<<<cg, 256, 0, stream>>>(actA, actB, Apack + 0 * APACK_PER_LAYER,
                                      alpha + 0 * 64, beta + 0 * 64, 1, 0);
    conv_wmma<<<cg, 256, 0, stream>>>(actB, actA, Apack + 1 * APACK_PER_LAYER,
                                      alpha + 1 * 64, beta + 1 * 64, 0, 0);
    conv_wmma<<<cg, 256, 0, stream>>>(actA, actB, Apack + 2 * APACK_PER_LAYER,
                                      alpha + 2 * 64, beta + 2 * 64, 0, 0);
    conv_wmma<<<cg, 256, 0, stream>>>(actB, actA, Apack + 3 * APACK_PER_LAYER,
                                      alpha + 3 * 64, beta + 3 * 64, 0, 0);
    conv_wmma<<<cg, 256, 0, stream>>>(actA, actB, Apack + 4 * APACK_PER_LAYER,
                                      alpha + 4 * 64, beta + 4 * 64, 0, 0);
    conv_wmma<<<cg, 256, 0, stream>>>(actB, actA, Apack + 5 * APACK_PER_LAYER,
                                      alpha + 5 * 64, beta + 5 * 64, 0, 0);
    conv_wmma<<<cg, 256, 0, stream>>>(actA, gen,  Apack + 6 * APACK_PER_LAYER,
                                      alpha + 6 * 64, beta + 6 * 64, 1, 1);

    fused_einsum<<<dim3(HW / 16, HW / 16, NB), 256, 0, stream>>>(
        input, gen, bases, (float*)d_out);
}
